// DSSMOne_2534030705315
// MI455X (gfx1250) — matrix operand order repaired and checked by reading, not compiled, not run on hardware
//
#include <hip/hip_runtime.h>
#include <cmath>

// ---------------- problem constants ----------------
constexpr int Bsz = 512, QL = 64, DL = 256, H = 256, CO = 512, KS = 3;
constexpr int KMAX = 3, LAT = 128;
constexpr int KCONV = H * KS;              // 768 = im2col K
constexpr int QLP = QL + 2, DLP = DL + 2;  // padded activation rows
constexpr int QMP = 64, DMP = 256;         // padded output lengths (mult of 64)
constexpr int QLOUT = QL - KS + 1;         // 62
constexpr int DLOUT = DL - KS + 1;         // 254

typedef __attribute__((ext_vector_type(16))) __bf16 v16bf;
typedef __attribute__((ext_vector_type(8)))  float  v8f;
typedef __attribute__((ext_vector_type(4)))  int    v4i;

#if __has_builtin(__builtin_amdgcn_global_load_async_to_lds_b128) && \
    __has_builtin(__builtin_amdgcn_s_wait_asynccnt)
#define USE_ASYNC_LDS 1
#else
#define USE_ASYNC_LDS 0
#endif

#if USE_ASYNC_LDS
__device__ __forceinline__ void async_cp16(const __bf16* g, __bf16* l) {
  __builtin_amdgcn_global_load_async_to_lds_b128(
      (__attribute__((address_space(1))) v4i*)(void*)g,
      (__attribute__((address_space(3))) v4i*)(void*)l, 0, 0);
}
#endif

// ---------------- prep kernels ----------------
// wT[co][dk*H + h] = w[co][h][dk]  (bf16)  -> B operand rows are K-contiguous
__global__ void conv_w_transpose(const float* __restrict__ w, __bf16* __restrict__ wT) {
  int i = blockIdx.x * blockDim.x + threadIdx.x;
  if (i >= CO * KCONV) return;
  int co = i / KCONV, kk = i - co * KCONV;
  int dk = kk >> 8, h = kk & 255;              // kk = dk*256 + h
  wT[i] = (__bf16)w[(co * H + h) * KS + dk];
}

__global__ void f32_to_bf16_copy(const float* __restrict__ s, __bf16* __restrict__ d, int n) {
  int i = blockIdx.x * blockDim.x + threadIdx.x;
  if (i < n) d[i] = (__bf16)s[i];
}

// x[b][t][h] = bf16(emb[tok[b][t]][h]); zero pad rows t in [L, Lpad)
__global__ void embed_bf16(const int* __restrict__ tok, const float* __restrict__ emb,
                           __bf16* __restrict__ x, int L, int Lpad) {
  const int b = blockIdx.x, t = blockIdx.y, h = threadIdx.x;
  __bf16 v = (__bf16)0.0f;
  if (t < L) v = (__bf16)emb[(long long)tok[b * L + t] * H + h];
  x[((long long)b * Lpad + t) * H + h] = v;
}

// ---------------- WMMA GEMM ----------------
// D[m][n] = sum_k A[m][k] * Bt[n][k] + bias[n]
// A: bf16, row stride lda (rows may overlap: conv im2col), per-z offset aStrideZ
// Bt: bf16 [Ntot][Ktot], N-major, K-contiguous.  Ktot % 64 == 0.
// storeTanhF32: 0 -> store bf16 raw, 1 -> store f32 tanh
__global__ __launch_bounds__(256)
void gemm_wmma_bf16(const __bf16* __restrict__ A, int lda, long long aStrideZ,
                    const __bf16* __restrict__ Bt, int Ktot,
                    const float* __restrict__ bias,
                    void* __restrict__ Cout, long long cStrideZ, int ldc,
                    int Mvalid, int Mblocks, int storeTanhF32) {
  // K-step 64; rows padded 64->72 elems (36 banks, conflict-free b128 reads)
  __shared__ __bf16 ldsA[2][64 * 72];    // 2 x 9 KB
  __shared__ __bf16 ldsB[2][128 * 72];   // 2 x 18 KB

  const int tid  = threadIdx.x;
  const int z    = blockIdx.y;
  const int mb   = blockIdx.x % Mblocks;
  const int nb   = blockIdx.x / Mblocks;
  const int lane = tid & 31, wv = tid >> 5;
  const int lo   = lane & 15, hi = lane >> 4;
  const int m0w  = (wv & 3) * 16;     // 4 waves tile M
  const int n0w  = (wv >> 2) * 64;    // 2 waves tile N, each owns 4x16 columns

  const __bf16* Az = A  + (long long)z * aStrideZ + (long long)(mb * 64) * lda;
  const __bf16* Bn = Bt + (long long)(nb * 128) * Ktot;

  // cooperative-copy chunk coords: 16B chunks, 8 per 128B row
  const int cRow = tid >> 3, cC8 = (tid & 7) * 8;

  v8f acc[4] = {};

  const int steps = Ktot >> 6;

#if USE_ASYNC_LDS
  auto issueTile = [&](int buf, int kt) {
    // A: 64 rows x 64 elems = 512 chunks (2 per thread)
    async_cp16(Az + (long long)cRow * lda + kt + cC8,        &ldsA[buf][cRow * 72 + cC8]);
    async_cp16(Az + (long long)(cRow + 32) * lda + kt + cC8, &ldsA[buf][(cRow + 32) * 72 + cC8]);
    // B: 128 rows x 64 elems = 1024 chunks (4 per thread)
#pragma unroll
    for (int r = 0; r < 4; ++r) {
      int row = cRow + r * 32;
      async_cp16(Bn + (long long)row * Ktot + kt + cC8, &ldsB[buf][row * 72 + cC8]);
    }
  };
#else
  auto fetchTile = [&](int kt, uint4* aR, uint4* bR) {
    aR[0] = *(const uint4*)(Az + (long long)cRow * lda + kt + cC8);
    aR[1] = *(const uint4*)(Az + (long long)(cRow + 32) * lda + kt + cC8);
#pragma unroll
    for (int r = 0; r < 4; ++r)
      bR[r] = *(const uint4*)(Bn + (long long)(cRow + r * 32) * Ktot + kt + cC8);
  };
  auto stashTile = [&](int buf, const uint4* aR, const uint4* bR) {
    *(uint4*)(&ldsA[buf][cRow * 72 + cC8])        = aR[0];
    *(uint4*)(&ldsA[buf][(cRow + 32) * 72 + cC8]) = aR[1];
#pragma unroll
    for (int r = 0; r < 4; ++r)
      *(uint4*)(&ldsB[buf][(cRow + r * 32) * 72 + cC8]) = bR[r];
  };
#endif

  auto compute = [&](int buf) {
    const __bf16* bufA = ldsA[buf];
    const __bf16* bufB = ldsB[buf];
#pragma unroll
    for (int kf = 0; kf < 2; ++kf) {
      // A fragment: lane lo = M row; elem e -> K = e + 16*hi (16 contiguous)
      union { uint4 u[2]; v16bf v; } fa;
      const __bf16* pa = &bufA[(m0w + lo) * 72 + kf * 32 + hi * 16];
      fa.u[0] = *(const uint4*)(pa);
      fa.u[1] = *(const uint4*)(pa + 8);
#pragma unroll
      for (int j = 0; j < 4; ++j) {
        // B fragment: lane lo = N col; elem e -> K = e + 16*hi (16 contiguous)
        union { uint4 u[2]; v16bf v; } fb;
        const __bf16* pb = &bufB[(n0w + j * 16 + lo) * 72 + kf * 32 + hi * 16];
        fb.u[0] = *(const uint4*)(pb);
        fb.u[1] = *(const uint4*)(pb + 8);
        acc[j] = __builtin_amdgcn_wmma_f32_16x16x32_bf16(
            false, fa.v, false, fb.v, (short)0, acc[j], false, false);
      }
    }
  };

  // ---- preload tile 0 into buffer 0 ----
#if USE_ASYNC_LDS
  issueTile(0, 0);
  __builtin_amdgcn_s_wait_asynccnt(0);
#else
  {
    uint4 aR[2], bR[4];
    fetchTile(0, aR, bR);
    stashTile(0, aR, bR);
  }
#endif
  __syncthreads();

  // ---- main loop: one barrier per K-step, double-buffered ----
  for (int it = 0; it < steps; ++it) {
    const int cur = it & 1, nxt = cur ^ 1;
#if USE_ASYNC_LDS
    if (it + 1 < steps) issueTile(nxt, (it + 1) << 6);   // in flight during compute
    compute(cur);
    if (it + 1 < steps) __builtin_amdgcn_s_wait_asynccnt(0);
#else
    uint4 aR[2], bR[4];
    if (it + 1 < steps) fetchTile((it + 1) << 6, aR, bR);  // loads in flight
    compute(cur);
    if (it + 1 < steps) stashTile(nxt, aR, bR);
#endif
    __syncthreads();
  }

  // epilogue: C/D layout -> lane lo = N, reg r -> M = r + 8*hi
#pragma unroll
  for (int j = 0; j < 4; ++j) {
    const int n = nb * 128 + n0w + j * 16 + lo;
    const float bv = bias[n];
#pragma unroll
    for (int r = 0; r < 8; ++r) {
      const int m = mb * 64 + m0w + r + 8 * hi;
      if (m < Mvalid) {
        float val = acc[j][r] + bv;
        long long o = (long long)z * cStrideZ + (long long)m * ldc + n;
        if (storeTanhF32) ((float*)Cout)[o] = tanhf(val);
        else              ((__bf16*)Cout)[o] = (__bf16)val;
      }
    }
  }
}

// ---------------- k-max(3) pool + tanh ----------------
// c: bf16 [B][Mpad][CO]; out: bf16 [B][KMAX][CO] (rows ordered by ascending index)
__global__ void kmax_pool_tanh(const __bf16* __restrict__ c, int Mpad, int Lout,
                               __bf16* __restrict__ kout) {
  const int b  = blockIdx.x;
  const int co = blockIdx.y * blockDim.x + threadIdx.x;
  const __bf16* p = c + (long long)b * Mpad * CO + co;
  float v1 = -1e30f, v2 = -1e30f, v3 = -1e30f;
  int   i1 = 0, i2 = 0, i3 = 0;
  for (int t = 0; t < Lout; ++t) {
    float v = (float)p[(long long)t * CO];
    if (v > v1)      { v3 = v2; i3 = i2; v2 = v1; i2 = i1; v1 = v; i1 = t; }
    else if (v > v2) { v3 = v2; i3 = i2; v2 = v;  i2 = t; }
    else if (v > v3) { v3 = v;  i3 = t; }
  }
  // sort the 3 picks by ascending index
  if (i1 > i2) { int ti = i1; i1 = i2; i2 = ti; float tv = v1; v1 = v2; v2 = tv; }
  if (i2 > i3) { int ti = i2; i2 = i3; i3 = ti; float tv = v2; v2 = v3; v3 = tv; }
  if (i1 > i2) { int ti = i1; i1 = i2; i2 = ti; float tv = v1; v1 = v2; v2 = tv; }
  long long base = (long long)b * KMAX * CO + co;
  kout[base]           = (__bf16)tanhf(v1);
  kout[base + CO]      = (__bf16)tanhf(v2);
  kout[base + 2 * CO]  = (__bf16)tanhf(v3);
}

// ---------------- final 768 -> 1 ----------------
__global__ void gamma_out(const float* __restrict__ sq, const float* __restrict__ sd,
                          const float* __restrict__ gw, const float* __restrict__ gb,
                          float* __restrict__ out) {
  const int b = blockIdx.x, tid = threadIdx.x;
  const int HALF = KMAX * LAT;  // 384
  float s = 0.f;
  for (int i = tid; i < 2 * HALF; i += 256) {
    float x = (i < HALF) ? sq[b * HALF + i] : sd[b * HALF + (i - HALF)];
    s += x * gw[i];
  }
  __shared__ float red[256];
  red[tid] = s;
  __syncthreads();
  for (int off = 128; off > 0; off >>= 1) {
    if (tid < off) red[tid] += red[tid + off];
    __syncthreads();
  }
  if (tid == 0) out[b] = red[0] + gb[0];
}

// ---------------- launcher ----------------
extern "C" void kernel_launch(void* const* d_in, const int* in_sizes, int n_in,
                              void* d_out, int out_size, void* d_ws, size_t ws_size,
                              hipStream_t stream) {
  const int*   query   = (const int*)d_in[0];
  const int*   doc     = (const int*)d_in[1];
  const float* emb     = (const float*)d_in[2];
  const float* qconv_w = (const float*)d_in[3];
  const float* qconv_b = (const float*)d_in[4];
  const float* dconv_w = (const float*)d_in[5];
  const float* dconv_b = (const float*)d_in[6];
  const float* qsem_w  = (const float*)d_in[7];
  const float* qsem_b  = (const float*)d_in[8];
  const float* dsem_w  = (const float*)d_in[9];
  const float* dsem_b  = (const float*)d_in[10];
  const float* gamma_w = (const float*)d_in[11];  // flat [768]
  const float* gamma_b = (const float*)d_in[12];

  char* ws = (char*)d_ws;
  size_t off = 0;
  auto take = [&](size_t bytes) -> char* {
    off = (off + 255) & ~(size_t)255;
    char* p = ws + off;
    off += bytes;
    return p;
  };
  __bf16* xq    = (__bf16*)take((size_t)Bsz * QLP * H * 2);
  __bf16* xd    = (__bf16*)take((size_t)Bsz * DLP * H * 2);
  __bf16* qwT   = (__bf16*)take((size_t)CO * KCONV * 2);
  __bf16* dwT   = (__bf16*)take((size_t)CO * KCONV * 2);
  __bf16* qsemT = (__bf16*)take((size_t)LAT * CO * 2);
  __bf16* dsemT = (__bf16*)take((size_t)LAT * CO * 2);
  __bf16* convq = (__bf16*)take((size_t)Bsz * QMP * CO * 2);
  __bf16* convd = (__bf16*)take((size_t)Bsz * DMP * CO * 2);
  __bf16* kq    = (__bf16*)take((size_t)Bsz * KMAX * CO * 2);
  __bf16* kd    = (__bf16*)take((size_t)Bsz * KMAX * CO * 2);
  float*  sq    = (float*)take((size_t)Bsz * KMAX * LAT * 4);
  float*  sd    = (float*)take((size_t)Bsz * KMAX * LAT * 4);

  // weight prep
  conv_w_transpose<<<(CO * KCONV + 255) / 256, 256, 0, stream>>>(qconv_w, qwT);
  conv_w_transpose<<<(CO * KCONV + 255) / 256, 256, 0, stream>>>(dconv_w, dwT);
  f32_to_bf16_copy<<<(LAT * CO + 255) / 256, 256, 0, stream>>>(qsem_w, qsemT, LAT * CO);
  f32_to_bf16_copy<<<(LAT * CO + 255) / 256, 256, 0, stream>>>(dsem_w, dsemT, LAT * CO);

  // embedding gather -> bf16 [B, Lpad, H]
  embed_bf16<<<dim3(Bsz, QLP), H, 0, stream>>>(query, emb, xq, QL, QLP);
  embed_bf16<<<dim3(Bsz, DLP), H, 0, stream>>>(doc,   emb, xd, DL, DLP);

  // conv as WMMA GEMM: M = Lout (padded), N = CO, K = 768; A row stride = H (im2col)
  gemm_wmma_bf16<<<dim3((QMP / 64) * (CO / 128), Bsz), 256, 0, stream>>>(
      xq, H, (long long)QLP * H, qwT, KCONV, qconv_b,
      convq, (long long)QMP * CO, CO, QLOUT, QMP / 64, 0);
  gemm_wmma_bf16<<<dim3((DMP / 64) * (CO / 128), Bsz), 256, 0, stream>>>(
      xd, H, (long long)DLP * H, dwT, KCONV, dconv_b,
      convd, (long long)DMP * CO, CO, DLOUT, DMP / 64, 0);

  // k-max(3) + tanh -> bf16 [B,3,CO] (A matrix of sem GEMM)
  kmax_pool_tanh<<<dim3(Bsz, CO / 256), 256, 0, stream>>>(convq, QMP, QLOUT, kq);
  kmax_pool_tanh<<<dim3(Bsz, CO / 256), 256, 0, stream>>>(convd, DMP, DLOUT, kd);

  // sem layer as WMMA GEMM: M = B*3 = 1536, K = 512, N = 128; tanh f32 out
  gemm_wmma_bf16<<<dim3((Bsz * KMAX / 64) * (LAT / 128), 1), 256, 0, stream>>>(
      kq, CO, 0, qsemT, CO, qsem_b, sq, 0, LAT, Bsz * KMAX, Bsz * KMAX / 64, 1);
  gemm_wmma_bf16<<<dim3((Bsz * KMAX / 64) * (LAT / 128), 1), 256, 0, stream>>>(
      kd, CO, 0, dsemT, CO, dsem_b, sd, 0, LAT, Bsz * KMAX, Bsz * KMAX / 64, 1);

  // final 768 -> 1
  gamma_out<<<Bsz, 256, 0, stream>>>(sq, sd, gamma_w, gamma_b, (float*)d_out);
}
